// QFIAttentionHead_39848706573833
// MI455X (gfx1250) — matrix operand order repaired and checked by reading, not compiled, not run on hardware
//
#include <hip/hip_runtime.h>
#include <hip/hip_bf16.h>
#include <math.h>

// Problem constants (from reference setup_inputs)
constexpr int B = 8, S = 2048, D = 256, H = 64;
constexpr float EPS = 1e-10f;

typedef __attribute__((ext_vector_type(16))) _Float16 v16h;
typedef __attribute__((ext_vector_type(8)))  _Float16 v8h;
typedef __attribute__((ext_vector_type(8)))  float    v8f;
typedef __attribute__((ext_vector_type(2)))  float    v2f;
typedef __attribute__((ext_vector_type(4)))  int      v4i;

#if __has_builtin(__builtin_amdgcn_global_load_async_to_lds_b128)
#define HAVE_ASYNC_LDS 1
#else
#define HAVE_ASYNC_LDS 0
#endif

// Abramowitz-Stegun 4.4.45 acos approximation (~7e-5 abs error) — far below
// the f16 quantization already present in the score matmul; replaces the
// large libm polynomial in the S^2 hot loop.
__device__ __forceinline__ float fast_acos(float x)
{
    const float ax = fabsf(x);
    float p = fmaf(ax, -0.0187293f, 0.0742610f);
    p = fmaf(p, ax, -0.2121144f);
    p = fmaf(p, ax, 1.5707288f);
    const float r = p * sqrtf(1.0f - ax);
    return (x < 0.0f) ? (3.14159265f - r) : r;
}

// ---------------------------------------------------------------------------
// Kernel 1: P = x @ W^T via V_WMMA_F32_16X16X4_F32 (full f32, K=4), then
// per-token simplex projection (bisection) and sqrt -> f16.
// One wave per (16-token tile, matrix). C tiles staged through 4 KB LDS.
// f32 A 16x4 layout (ISA 7.12.2): lane L row M=L&15, VGPR r -> K = (L>=16?2:0)+r
// f32 B 4x16 layout:              lane L col N=L&15, VGPR r -> K = (L>=16?2:0)+r
// ---------------------------------------------------------------------------
__global__ __launch_bounds__(32)
void proj_simplex_kernel(const float* __restrict__ x,
                         const float* __restrict__ Wq,
                         const float* __restrict__ Wk,
                         const float* __restrict__ Wv,
                         _Float16* __restrict__ qh,
                         _Float16* __restrict__ kh,
                         _Float16* __restrict__ vh)
{
    __shared__ float pt[16 * 64];               // one 16-token x 64-head tile

    const int tile = blockIdx.x;                // [0, B*S/16)
    const int mat  = blockIdx.y;                // 0..2  (q, k, v)
    const float* W  = (mat == 0) ? Wq : (mat == 1) ? Wk : Wv;
    _Float16*    Ot = (mat == 0) ? qh : (mat == 1) ? kh : vh;

    const int lane = threadIdx.x;
    const int m    = lane & 15;
    const int koff = (lane >> 4) * 2;           // K-pair held by this lane
    const int tok0 = tile * 16;

    v8f acc[4];
    #pragma unroll
    for (int ht = 0; ht < 4; ++ht) acc[ht] = (v8f){0,0,0,0,0,0,0,0};

    const float* xrow = x + (size_t)(tok0 + m) * D + koff;

    #pragma unroll 1
    for (int kc = 0; kc < D; kc += 4) {
        const v2f a = *(const v2f*)(xrow + kc);
        #pragma unroll
        for (int ht = 0; ht < 4; ++ht) {
            const v2f bw = *(const v2f*)(W + (size_t)(ht * 16 + m) * D + kc + koff);
            acc[ht] = __builtin_amdgcn_wmma_f32_16x16x4_f32(
                          false, a, false, bw, (short)0, acc[ht], false, false);
        }
    }

    // C layout: VGPR r -> row = r + (lane>=16 ? 8 : 0), col = lane&15
    #pragma unroll
    for (int ht = 0; ht < 4; ++ht)
        #pragma unroll
        for (int r = 0; r < 8; ++r)
            pt[(r + (lane >> 4) * 8) * 64 + ht * 16 + m] = acc[ht][r];
    __syncthreads();

    // Phase 2: lanes 0..15 each own one token's simplex projection.
    if (lane < 16) {
        float v[H];
        #pragma unroll
        for (int h = 0; h < H; ++h) v[h] = pt[lane * 64 + h];

        float vmax = -1e30f, vmin = 1e30f;
        #pragma unroll
        for (int h = 0; h < H; ++h) {
            vmax = fmaxf(vmax, v[h]);
            vmin = fminf(vmin, v[h]);
        }
        float lo = -vmax;                       // f(lo) = 0
        float hi = 1.0f / H - vmin;             // f(hi) >= 1
        #pragma unroll 1
        for (int it = 0; it < 32; ++it) {
            const float lam = 0.5f * (lo + hi);
            float s = 0.0f;
            #pragma unroll
            for (int h = 0; h < H; ++h)
                s += fmaxf(v[h] + lam, 0.0f);
            if (s < 1.0f) lo = lam; else hi = lam;
        }
        const float lam = 0.5f * (lo + hi);

        _Float16* dst = Ot + (size_t)(tok0 + lane) * H;
        #pragma unroll
        for (int h = 0; h < H; ++h) {
            const float p = fmaxf(v[h] + lam, 0.0f);
            dst[h] = (_Float16)sqrtf(fmaxf(p, EPS));
        }
    }
}

// ---------------------------------------------------------------------------
// Kernel 2: fused flash-style Fisher-Rao attention.
// One wave per (batch, 16-query tile). Streams 32-key tiles:
//   scores (2x 16x16)  via v_wmma_f32_16x16x32_f16 (K=H=64, 2 chained)
//   w = exp(-4*fast_acos(clip(score)))
//   acc += w(16x32) @ sqrt_v(32x64)  via 4x v_wmma_f32_16x16x32_f16
// V tiles staged with GLOBAL_LOAD_ASYNC_TO_LDS_B128 (ASYNCcnt) when available.
// Weight row-normalization cancels under the final L2 norm: out = acc^2/Σacc^2.
// ---------------------------------------------------------------------------
__global__ __launch_bounds__(32)
void fr_attention_kernel(const _Float16* __restrict__ qh,
                         const _Float16* __restrict__ kh,
                         const _Float16* __restrict__ vh,
                         float* __restrict__ out)
{
    __shared__ _Float16 sw[16 * 32];            // restage scores -> A operand
    __shared__ _Float16 sv[32 * 64];            // staged V key-tile

    const int b    = blockIdx.y;
    const int q0   = blockIdx.x * 16;
    const int lane = threadIdx.x;               // 0..31 (wave32)
    const int m    = lane & 15;
    const int khf  = lane >> 4;

    const _Float16* qb = qh + (size_t)b * S * H;
    const _Float16* kb = kh + (size_t)b * S * H;
    const _Float16* vb = vh + (size_t)b * S * H;

    // Query A-fragments (this code's contiguous k order, same as B): j -> K=khf*16+j
    v16h aq0, aq1;
    {
        const _Float16* qr = qb + (size_t)(q0 + m) * H;
        #pragma unroll
        for (int j = 0; j < 8; ++j) {
            aq0[j]     = qr[khf * 16 + j];
            aq0[j + 8] = qr[khf * 16 + 8 + j];
            aq1[j]     = qr[32 + khf * 16 + j];
            aq1[j + 8] = qr[32 + khf * 16 + 8 + j];
        }
    }

    v8f accv[4];
    #pragma unroll
    for (int ht = 0; ht < 4; ++ht) accv[ht] = (v8f){0,0,0,0,0,0,0,0};

    #pragma unroll 1
    for (int t = 0; t < S; t += 32) {
        if (t + 32 < S) {                       // global_prefetch_b8
            __builtin_prefetch(kb + (size_t)(t + 32) * H, 0, 1);
            __builtin_prefetch(vb + (size_t)(t + 32) * H, 0, 1);
        }

        // Stage V tile (32 keys x 64 heads, contiguous 4 KB). Async when possible
        // so the copy overlaps the score WMMAs + elementwise below.
        {
            const _Float16* vt = vb + (size_t)t * H;
#if HAVE_ASYNC_LDS
            #pragma unroll
            for (int c = 0; c < 8; ++c)
                __builtin_amdgcn_global_load_async_to_lds_b128(
                    (__attribute__((address_space(1))) v4i*)(vt + c * 256 + lane * 8),
                    (__attribute__((address_space(3))) v4i*)&sv[c * 256 + lane * 8],
                    0, 0);
#else
            #pragma unroll
            for (int c = 0; c < 8; ++c)
                *(v8h*)&sv[c * 256 + lane * 8] =
                    *(const v8h*)&vt[c * 256 + lane * 8];
#endif
        }

        // --- Scores for two 16-key tiles (contraction over H=64) ---
        v8f c0 = (v8f){0,0,0,0,0,0,0,0};
        v8f c1 = (v8f){0,0,0,0,0,0,0,0};
        #pragma unroll
        for (int tile = 0; tile < 2; ++tile) {
            // B 32x16 f16, contiguous k order matching A: lane col N=lane&15, halves j -> K = khf*16+j
            const _Float16* kr = kb + (size_t)(t + tile * 16 + m) * H;
            v16h bk0, bk1;
            #pragma unroll
            for (int j = 0; j < 16; ++j) {
                bk0[j] = kr[khf * 16 + j];          // h = 0..31
                bk1[j] = kr[32 + khf * 16 + j];     // h = 32..63
            }
            v8f c = (tile == 0) ? c0 : c1;
            c = __builtin_amdgcn_wmma_f32_16x16x32_f16(
                    false, aq0, false, bk0, (short)0, c, false, false);
            c = __builtin_amdgcn_wmma_f32_16x16x32_f16(
                    false, aq1, false, bk1, (short)0, c, false, false);
            if (tile == 0) c0 = c; else c1 = c;
        }

        // --- w = exp(-4*acos(clip(inner))) ; restage f16 into sw ---
        #pragma unroll
        for (int r = 0; r < 8; ++r) {
            const int row = r + khf * 8;            // C layout row
            float i0 = fminf(fmaxf(c0[r], -1.0f + EPS), 1.0f - EPS);
            float i1 = fminf(fmaxf(c1[r], -1.0f + EPS), 1.0f - EPS);
            sw[row * 32 + m]      = (_Float16)__expf(-4.0f * fast_acos(i0));
            sw[row * 32 + 16 + m] = (_Float16)__expf(-4.0f * fast_acos(i1));
        }
        __syncthreads();

#if HAVE_ASYNC_LDS
#if __has_builtin(__builtin_amdgcn_s_wait_asynccnt)
        __builtin_amdgcn_s_wait_asynccnt(0);
#else
        asm volatile("s_wait_asynccnt 0" ::: "memory");
#endif
#endif

        // --- A fragment of w (16 queries x 32 keys) ---
        v16h aw;
        #pragma unroll
        for (int j = 0; j < 8; ++j) {
            aw[j]     = sw[m * 32 + khf * 16 + j];
            aw[j + 8] = sw[m * 32 + khf * 16 + 8 + j];
        }

        // --- acc[16q x 16h] += w @ sqrt_v for 4 head tiles (K=32 keys) ---
        #pragma unroll
        for (int ht = 0; ht < 4; ++ht) {
            v16h bv;                                 // B: K = khf*16+j keys
            #pragma unroll
            for (int j = 0; j < 16; ++j)
                bv[j] = sv[(khf * 16 + j) * 64 + ht * 16 + m];
            accv[ht] = __builtin_amdgcn_wmma_f32_16x16x32_f16(
                    false, aw, false, bv, (short)0, accv[ht], false, false);
        }
        __syncthreads();                             // protect sv/sw reuse
    }

    // --- out = acc^2 / sum_h acc^2 (weight & L2 norms cancel) ---
    float pn[8];
    #pragma unroll
    for (int r = 0; r < 8; ++r) {
        float s = 0.0f;
        #pragma unroll
        for (int ht = 0; ht < 4; ++ht) {
            const float tv = accv[ht][r];
            s += tv * tv;
        }
        #pragma unroll
        for (int msk = 1; msk < 16; msk <<= 1)
            s += __shfl_xor(s, msk, 16);             // reduce 16 columns
        pn[r] = s + 1e-20f;
    }

    float* ob = out + ((size_t)b * S + q0) * H;
    #pragma unroll
    for (int ht = 0; ht < 4; ++ht) {
        #pragma unroll
        for (int r = 0; r < 8; ++r) {
            const int row = r + khf * 8;
            const float v = accv[ht][r];
            ob[(size_t)row * H + ht * 16 + m] = v * v / pn[r];
        }
    }
}

// ---------------------------------------------------------------------------
extern "C" void kernel_launch(void* const* d_in, const int* in_sizes, int n_in,
                              void* d_out, int out_size, void* d_ws, size_t ws_size,
                              hipStream_t stream)
{
    const float* x  = (const float*)d_in[0];
    const float* Wq = (const float*)d_in[1];
    const float* Wk = (const float*)d_in[2];
    const float* Wv = (const float*)d_in[3];

    _Float16* qh = (_Float16*)d_ws;                      // 3 x 2 MB f16
    _Float16* kh = qh + (size_t)B * S * H;
    _Float16* vh = kh + (size_t)B * S * H;

    proj_simplex_kernel<<<dim3((B * S) / 16, 3), dim3(32), 0, stream>>>(
        x, Wq, Wk, Wv, qh, kh, vh);

    fr_attention_kernel<<<dim3(S / 16, B), dim3(32), 0, stream>>>(
        qh, kh, vh, (float*)d_out);
}
